// NCEAverage_46093589021323
// MI455X (gfx1250) — matrix operand order repaired and checked by reading, not compiled, not run on hardware
//
#include <hip/hip_runtime.h>

// NCE Average for MI455X (gfx1250, wave32).
// d_in: x[256,128] f32, y[256] i32, idx[256,4096] i32, memory[1e6,128] f32
// d_out: scores[256,4096] f32
//
// Memory-bound gather (512 MB) -> dots done on WMMA f16 pipe (free at this
// arithmetic intensity). One wave computes a 16-k score tile for one b via
// 4x v_wmma_f32_16x16x32_f16 with x[b] broadcast across the N dimension.

typedef __attribute__((ext_vector_type(16))) _Float16 v16h;
typedef __attribute__((ext_vector_type(8)))  float    v8f;
typedef __attribute__((ext_vector_type(4)))  float    v4f;

#define NCE_B 256
#define NCE_K 4096
#define NCE_D 128
#define NCE_N 1000000
#define NCE_INV_T (1.0f / 0.07f)

#define TILES_PER_B   (NCE_K / 16)            // 256
#define TOTAL_TILES   (NCE_B * TILES_PER_B)   // 65536
#define WAVES_PER_BLK 8
#define BLK1          (TOTAL_TILES / WAVES_PER_BLK) // 8192

__global__ __launch_bounds__(256)
void nce_score_kernel(const float* __restrict__ x,
                      const int*   __restrict__ y,
                      const int*   __restrict__ idx,
                      const float* __restrict__ memory,
                      float*       __restrict__ out,
                      float*       __restrict__ partials)
{
    const int lane = threadIdx.x & 31;
    const int wave = threadIdx.x >> 5;
    const int tile = blockIdx.x * WAVES_PER_BLK + wave;
    const int b    = tile >> 8;                 // tile / (K/16)
    const int k0   = (tile & (TILES_PER_B - 1)) << 4;

    const int m  = lane & 15;   // row (M) within the 16-k tile
    const int hi = lane >> 4;   // lane half selects K sub-ranges

    // effective index: column 0 of idx is replaced by the positive y[b]
    const int k   = k0 + m;
    const int row = (k == 0) ? y[b] : idx[b * NCE_K + k];

    const float* __restrict__ rp = memory + (size_t)row * NCE_D;
    const float* __restrict__ xp = x + b * NCE_D;

    v8f acc = {};
#pragma unroll
    for (int c = 0; c < 4; ++c) {
        const int dbase = c * 32;
        // A (16x32 f16): lanes 0-15 hold K=0..7 & 16..23, lanes 16-31 K=8..15 & 24..31
        const int offA1 = dbase + hi * 8;
        const int offA2 = dbase + 16 + hi * 8;
        v4f a0 = *(const v4f*)(rp + offA1);
        v4f a1 = *(const v4f*)(rp + offA1 + 4);
        v4f a2 = *(const v4f*)(rp + offA2);
        v4f a3 = *(const v4f*)(rp + offA2 + 4);
        // B (32x16 f16), x broadcast to all 16 columns: lanes 0-15 hold K=0..7 & 16..23,
        // lanes 16-31 hold K=8..15 & 24..31 (same K order as A).
        const int offB1 = dbase + hi * 8;
        const int offB2 = dbase + 16 + hi * 8;
        v4f b0 = *(const v4f*)(xp + offB1);
        v4f b1 = *(const v4f*)(xp + offB1 + 4);
        v4f b2 = *(const v4f*)(xp + offB2);
        v4f b3 = *(const v4f*)(xp + offB2 + 4);

        v16h A, Bv;
#pragma unroll
        for (int j = 0; j < 4; ++j) {
            A[j]      = (_Float16)a0[j];
            A[4 + j]  = (_Float16)a1[j];
            A[8 + j]  = (_Float16)a2[j];
            A[12 + j] = (_Float16)a3[j];
            Bv[j]      = (_Float16)b0[j];
            Bv[4 + j]  = (_Float16)b1[j];
            Bv[8 + j]  = (_Float16)b2[j];
            Bv[12 + j] = (_Float16)b3[j];
        }
        acc = __builtin_amdgcn_wmma_f32_16x16x32_f16(
            /*neg_a=*/false, A, /*neg_b=*/false, Bv,
            /*c_mod=*/(short)0, acc, /*reuse_a=*/false, /*reuse_b=*/false);
    }

    // All columns of the D tile are identical (B was a broadcast), so lane 0
    // holds scores k0+0..7 across acc[0..7] and lane 16 holds k0+8..15.
    float e[8];
    float s = 0.f;
#pragma unroll
    for (int r = 0; r < 8; ++r) {
        float v = __expf(acc[r] * NCE_INV_T);
        e[r] = v;
        s += v;
    }

    __shared__ float red[2 * WAVES_PER_BLK];
    if (m == 0) {
        float* op = out + b * NCE_K + k0 + hi * 8;
        *(v4f*)(op)     = (v4f){e[0], e[1], e[2], e[3]};
        *(v4f*)(op + 4) = (v4f){e[4], e[5], e[6], e[7]};
        red[wave * 2 + hi] = s;
    }
    __syncthreads();
    if (threadIdx.x == 0) {
        float t = 0.f;
#pragma unroll
        for (int i = 0; i < 2 * WAVES_PER_BLK; ++i) t += red[i];
        partials[blockIdx.x] = t;
    }
}

__global__ __launch_bounds__(256)
void nce_reduce_kernel(const float* __restrict__ partials,
                       float* __restrict__ invZ)
{
    __shared__ float red[256];
    float s = 0.f;
    for (int i = threadIdx.x; i < BLK1; i += 256) s += partials[i];
    red[threadIdx.x] = s;
    __syncthreads();
    for (int off = 128; off > 0; off >>= 1) {
        if (threadIdx.x < off) red[threadIdx.x] += red[threadIdx.x + off];
        __syncthreads();
    }
    if (threadIdx.x == 0) {
        float S = red[0];
        float Z = S * ((float)NCE_N / (float)(NCE_B * NCE_K)); // mean * N
        *invZ = 1.0f / Z;
    }
}

__global__ __launch_bounds__(256)
void nce_scale_kernel(float* __restrict__ out, const float* __restrict__ invZ)
{
    const int i = blockIdx.x * 256 + threadIdx.x;
    out[i] *= *invZ;
}

extern "C" void kernel_launch(void* const* d_in, const int* in_sizes, int n_in,
                              void* d_out, int out_size, void* d_ws, size_t ws_size,
                              hipStream_t stream)
{
    const float* x      = (const float*)d_in[0];
    const int*   y      = (const int*)d_in[1];
    const int*   idx    = (const int*)d_in[2];
    const float* memory = (const float*)d_in[3];
    float* out = (float*)d_out;

    float* wsf      = (float*)d_ws;
    float* invZ     = wsf;        // 1 float
    float* partials = wsf + 16;   // BLK1 floats (aligned region)

    nce_score_kernel<<<BLK1, 256, 0, stream>>>(x, y, idx, memory, out, partials);
    nce_reduce_kernel<<<1, 256, 0, stream>>>(partials, invZ);
    nce_scale_kernel<<<(NCE_B * NCE_K) / 256, 256, 0, stream>>>(out, invZ);
}
